// AutoFormer_55216099557493
// MI455X (gfx1250) — compile-verified, matrix operand order checked by reading, not run on hardware
//
#include <hip/hip_runtime.h>

// ---------------------------------------------------------------------------
// Autoformer forward for gfx1250 (MI455X): all GEMMs via v_wmma_f32_16x16x32_bf16
// Block tile 128x64, 8 wave32 waves (each 32x32 of C), double-buffered LDS,
// software-pipelined staging, weights pre-converted to bf16 [N,K] layout.
// ---------------------------------------------------------------------------

typedef __attribute__((ext_vector_type(16))) __bf16          v16bf;
typedef __attribute__((ext_vector_type(8)))  float           v8f;
typedef __attribute__((ext_vector_type(4)))  unsigned short  us4;
typedef __attribute__((ext_vector_type(8)))  unsigned short  us8;

#define BB   32
#define SS   512
#define INN  256
#define DD   512
#define HH   8
#define LL   2
#define DFF  2048
#define NT   424
#define HALF 12   // K=25 -> K//2

#define ACT_NONE 0
#define ACT_GELU 1
#define ACT_RELU 2

#define LDT  48            // ushort stride per tile row (96B: keeps 32B frag alignment)
#define ASZ  (128 * LDT)   // A tile: 128 rows x 32 k
#define BSZ  (64 * LDT)    // B tile: 64 n x 32 k (already K-contiguous in memory)

// round-to-nearest-even fp32 -> bf16 bits
__device__ __forceinline__ unsigned short f2bf(float f) {
    unsigned int u = __float_as_uint(f);
    u += 0x7fffu + ((u >> 16) & 1u);
    return (unsigned short)(u >> 16);
}

__device__ __forceinline__ float gelu_f(float x) {
    float x3 = x * x * x;
    float t  = tanhf(0.7978845608028654f * (x + 0.044715f * x3));
    return 0.5f * x * (1.0f + t);
}

// Registers holding one thread's share of a (128x32 A, 64x32 B) tile pair.
struct TileRegs {
    float4 a[4];   // A: 4 x float4 (fp32, to be converted)
    us8    b;      // B: one 16B chunk of bf16 (already converted+transposed)
};

// Phase 1: issue all global loads (no consumption -> loads pipeline).
template <bool GUARD>
__device__ __forceinline__ TileRegs stage_load(
    const float* __restrict__ A, const unsigned short* __restrict__ Bt,
    int M, int N, int K, int blockM, int blockN, int k0, int tid)
{
    TileRegs t;
#pragma unroll
    for (int i = 0; i < 4; ++i) {
        int f  = tid + i * 256;            // 0..1023
        int r  = f >> 3;                   // 0..127
        int c4 = (f & 7) * 4;              // 0,4,..,28
        int gr = blockM + r;
        if (GUARD && gr >= M) gr = M - 1;
        t.a[i] = *(const float4*)&A[(size_t)gr * K + k0 + c4];
    }
    {
        int nn = tid >> 2;                 // 0..63
        int kc = (tid & 3) * 8;            // 0,8,16,24 (ushorts)
        int gn = blockN + nn;
        if (GUARD && gn >= N) gn = N - 1;
        t.b = *(const us8*)&Bt[(size_t)gn * K + k0 + kc];
    }
    return t;
}

// Phase 2: convert (A only) and store to LDS.
template <bool GUARD>
__device__ __forceinline__ void stage_store(
    const TileRegs& t, int M, int N, int blockM, int blockN, int tid,
    unsigned short* __restrict__ As, unsigned short* __restrict__ Bs)
{
#pragma unroll
    for (int i = 0; i < 4; ++i) {
        int f  = tid + i * 256;
        int r  = f >> 3;
        int c4 = (f & 7) * 4;
        float4 v = t.a[i];
        if (GUARD && (blockM + r) >= M) { v.x = 0.f; v.y = 0.f; v.z = 0.f; v.w = 0.f; }
        us4 p; p.x = f2bf(v.x); p.y = f2bf(v.y); p.z = f2bf(v.z); p.w = f2bf(v.w);
        *(us4*)&As[r * LDT + c4] = p;      // 8B aligned
    }
    {
        int nn = tid >> 2;
        int kc = (tid & 3) * 8;
        us8 b = t.b;
        if (GUARD && (blockN + nn) >= N) b = (us8){0,0,0,0,0,0,0,0};
        *(us8*)&Bs[nn * LDT + kc] = b;     // 16B aligned
    }
}

// ---------------------------------------------------------------------------
// GEMM:  C[M,N] = act(A[M,K] @ W + bias) + resid,  W given as bf16 Bt[N,K].
// Row-major A/C. K multiple of 32. Grid: x over N/64, y over M/128.
// ---------------------------------------------------------------------------
template <int ACT, bool GUARD>
__global__ __launch_bounds__(256) void gemm_wmma_bf16(
    const float* __restrict__ A, const unsigned short* __restrict__ Bt,
    const float* __restrict__ bias, const float* __restrict__ resid,
    float* __restrict__ C, int M, int N, int K)
{
    __shared__ alignas(32) unsigned short As[2 * ASZ];   // double-buffered
    __shared__ alignas(32) unsigned short Bs[2 * BSZ];

    const int tid    = threadIdx.x;
    const int lane   = tid & 31;
    const int wave   = tid >> 5;                 // 0..7
    const int wm     = wave & 3;                 // M sub-tile (32 rows each)
    const int wn     = wave >> 2;                // N half (32 cols each)
    const int blockM = blockIdx.y * 128;
    const int blockN = blockIdx.x * 64;

    v8f acc[2][2];
#pragma unroll
    for (int a = 0; a < 2; ++a)
#pragma unroll
        for (int h = 0; h < 2; ++h)
            acc[a][h] = (v8f){0.f, 0.f, 0.f, 0.f, 0.f, 0.f, 0.f, 0.f};

    const int l16  = lane & 15;
    const int koff = (lane >> 4) * 16;
    const int nk   = K >> 5;

    // prologue: stage tile 0
    {
        TileRegs t0 = stage_load<GUARD>(A, Bt, M, N, K, blockM, blockN, 0, tid);
        stage_store<GUARD>(t0, M, N, blockM, blockN, tid, As, Bs);
    }
    __syncthreads();

    for (int kb = 0; kb < nk; ++kb) {
        const unsigned short* Ac = As + (kb & 1) * ASZ;
        const unsigned short* Bc = Bs + (kb & 1) * BSZ;

        // software pipeline: issue next tile's global loads before the WMMAs
        TileRegs tn;
        const bool have_next = (kb + 1 < nk);
        if (have_next)
            tn = stage_load<GUARD>(A, Bt, M, N, K, blockM, blockN, (kb + 1) * 32, tid);

        // fragments: lane l -> row/col (l&15), k = 16*(l>>4)+s (contiguous, same order for A and B)
        v16bf a0, a1, b0, b1;
        __builtin_memcpy(&a0, &Ac[(wm * 32 +      l16) * LDT + koff], 32);
        __builtin_memcpy(&a1, &Ac[(wm * 32 + 16 + l16) * LDT + koff], 32);
        __builtin_memcpy(&b0, &Bc[(wn * 32 +      l16) * LDT + koff], 32);
        __builtin_memcpy(&b1, &Bc[(wn * 32 + 16 + l16) * LDT + koff], 32);

        acc[0][0] = __builtin_amdgcn_wmma_f32_16x16x32_bf16(false, a0, false, b0, (short)0, acc[0][0], false, false);
        acc[0][1] = __builtin_amdgcn_wmma_f32_16x16x32_bf16(false, a0, false, b1, (short)0, acc[0][1], false, false);
        acc[1][0] = __builtin_amdgcn_wmma_f32_16x16x32_bf16(false, a1, false, b0, (short)0, acc[1][0], false, false);
        acc[1][1] = __builtin_amdgcn_wmma_f32_16x16x32_bf16(false, a1, false, b1, (short)0, acc[1][1], false, false);

        if (have_next)
            stage_store<GUARD>(tn, M, N, blockM, blockN, tid,
                               As + ((kb + 1) & 1) * ASZ, Bs + ((kb + 1) & 1) * BSZ);
        __syncthreads();
    }

    // epilogue: D layout — lane l holds col n=(l&15); VGPR r holds row r+8*(l>>4)
    const int rowOff = (lane >> 4) * 8;
#pragma unroll
    for (int a = 0; a < 2; ++a) {
#pragma unroll
        for (int r = 0; r < 8; ++r) {
            int gm = blockM + wm * 32 + a * 16 + rowOff + r;
#pragma unroll
            for (int h = 0; h < 2; ++h) {
                int   gn = blockN + wn * 32 + h * 16 + l16;
                float v  = acc[a][h][r];
                if (!GUARD || (gm < M && gn < N)) {
                    if (bias) v += bias[gn];
                    if (ACT == ACT_GELU) v = gelu_f(v);
                    if (ACT == ACT_RELU) v = fmaxf(v, 0.f);
                    if (resid) v += resid[(size_t)gm * N + gn];
                    C[(size_t)gm * N + gn] = v;
                }
            }
        }
    }
}

// ---------------------------------------------------------------------------
// Weight pre-pass: W[K,N] fp32 -> Wt[N,K] bf16 (transpose + convert).
// One-time cost; reads coalesced along N.
// ---------------------------------------------------------------------------
__global__ __launch_bounds__(256) void weight_tr_bf16_kernel(
    const float* __restrict__ W, unsigned short* __restrict__ Wt, int K, int N)
{
    int i = blockIdx.x * 256 + threadIdx.x;
    if (i >= K * N) return;
    int k = i / N, n = i % N;
    Wt[(size_t)n * K + k] = f2bf(W[i]);
}

// ---------------------------------------------------------------------------
// Autocorrelation + softmax + *V, one workgroup per (b, h, dk) channel.
// corr[n] = sum_t q[t] * k[(t-n) mod S]; out[s] = softmax(corr)[s] * v[s].
// Safe in place (out may alias q): all reads staged to LDS first.
// ---------------------------------------------------------------------------
__global__ __launch_bounds__(256) void autocorr_softmax_kernel(
    const float* __restrict__ q, const float* __restrict__ k,
    const float* __restrict__ v, float* __restrict__ out)
{
    __shared__ float sq[SS], sk[SS], sv[SS];
    __shared__ float red[256];

    const int ch = blockIdx.x;                // 0 .. B*D-1
    const int b  = ch / DD;
    const int d  = ch % DD;
    const size_t base = (size_t)b * SS * DD + d;
    const int t  = threadIdx.x;

    for (int s = t; s < SS; s += 256) {
        sq[s] = q[base + (size_t)s * DD];
        sk[s] = k[base + (size_t)s * DD];
        sv[s] = v[base + (size_t)s * DD];
    }
    __syncthreads();

    float cvals[2];
    float lmax = -3.0e38f;
#pragma unroll
    for (int i = 0; i < 2; ++i) {
        const int n = t + i * 256;
        float acc = 0.f;
        for (int tt = 0; tt < SS; ++tt)
            acc = fmaf(sq[tt], sk[(tt - n) & (SS - 1)], acc);
        cvals[i] = acc;
        lmax = fmaxf(lmax, acc);
    }
    red[t] = lmax; __syncthreads();
    for (int off = 128; off > 0; off >>= 1) {
        if (t < off) red[t] = fmaxf(red[t], red[t + off]);
        __syncthreads();
    }
    const float mx = red[0]; __syncthreads();

    float evals[2];
    float lsum = 0.f;
#pragma unroll
    for (int i = 0; i < 2; ++i) { evals[i] = __expf(cvals[i] - mx); lsum += evals[i]; }
    red[t] = lsum; __syncthreads();
    for (int off = 128; off > 0; off >>= 1) {
        if (t < off) red[t] += red[t + off];
        __syncthreads();
    }
    const float inv = 1.f / red[0];

#pragma unroll
    for (int i = 0; i < 2; ++i) {
        const int n = t + i * 256;
        out[base + (size_t)n * DD] = evals[i] * inv * sv[n];
    }
}

// ---------------------------------------------------------------------------
// Series decomposition: out = y - moving_avg(y, K=25, count_include_pad=False)
// ---------------------------------------------------------------------------
__global__ __launch_bounds__(256) void decomp_kernel(
    const float* __restrict__ y, float* __restrict__ out, int total)
{
    int i = blockIdx.x * 256 + threadIdx.x;
    if (i >= total) return;
    const int s = (i / DD) % SS;
    const size_t colBase = (size_t)i - (size_t)s * DD;   // index of (b, 0, d)
    int lo = s - HALF; if (lo < 0) lo = 0;
    int hi = s + HALF; if (hi > SS - 1) hi = SS - 1;
    float acc = 0.f;
    for (int j = lo; j <= hi; ++j)
        acc += y[colBase + (size_t)j * DD];
    out[i] = y[i] - acc / (float)(hi - lo + 1);
}

// ---------------------------------------------------------------------------
// Mean over sequence axis: [B,S,D] -> [B,D]
// ---------------------------------------------------------------------------
__global__ __launch_bounds__(256) void seqmean_kernel(
    const float* __restrict__ h, float* __restrict__ m)
{
    int i = blockIdx.x * 256 + threadIdx.x;    // 0 .. B*D-1
    if (i >= BB * DD) return;
    const int b = i / DD;
    const int d = i % DD;
    const size_t base = (size_t)b * SS * DD + d;
    float acc = 0.f;
    for (int s = 0; s < SS; ++s)
        acc += h[base + (size_t)s * DD];
    m[i] = acc * (1.0f / (float)SS);
}

// ---------------------------------------------------------------------------
// Host side
// ---------------------------------------------------------------------------
static void launch_gemm_full(int act, const float* A, const unsigned short* Bt,
                             const float* bias, const float* resid, float* C,
                             int M, int N, int K, hipStream_t stream)
{
    dim3 grid(N / 64, M / 128);
    if (act == ACT_GELU)
        gemm_wmma_bf16<ACT_GELU, false><<<grid, 256, 0, stream>>>(A, Bt, bias, resid, C, M, N, K);
    else
        gemm_wmma_bf16<ACT_NONE, false><<<grid, 256, 0, stream>>>(A, Bt, bias, resid, C, M, N, K);
}

static void launch_gemm_guard(int act, const float* A, const unsigned short* Bt,
                              const float* bias, const float* resid, float* C,
                              int M, int N, int K, hipStream_t stream)
{
    dim3 grid((N + 63) / 64, (M + 127) / 128);
    if (act == ACT_RELU)
        gemm_wmma_bf16<ACT_RELU, true><<<grid, 256, 0, stream>>>(A, Bt, bias, resid, C, M, N, K);
    else
        gemm_wmma_bf16<ACT_NONE, true><<<grid, 256, 0, stream>>>(A, Bt, bias, resid, C, M, N, K);
}

static void launch_wt(const float* W, unsigned short* Wt, int K, int N, hipStream_t stream)
{
    int total = K * N;
    weight_tr_bf16_kernel<<<(total + 255) / 256, 256, 0, stream>>>(W, Wt, K, N);
}

extern "C" void kernel_launch(void* const* d_in, const int* in_sizes, int n_in,
                              void* d_out, int out_size, void* d_ws, size_t ws_size,
                              hipStream_t stream)
{
    (void)in_sizes; (void)n_in; (void)out_size; (void)ws_size;

    const float* x        = (const float*)d_in[0];
    const float* embed_w  = (const float*)d_in[1];
    const float* embed_b  = (const float*)d_in[2];
    const float* qkvo_w   = (const float*)d_in[3];   // [L,4,D,D]
    const float* qkvo_b   = (const float*)d_in[4];   // [L,4,D]
    const float* ffn_w1   = (const float*)d_in[5];   // [L,D,DFF]
    const float* ffn_b1   = (const float*)d_in[6];   // [L,DFF]
    const float* ffn_w2   = (const float*)d_in[7];   // [L,DFF,D]
    const float* ffn_b2   = (const float*)d_in[8];   // [L,D]
    const float* proj_w1  = (const float*)d_in[9];
    const float* proj_b1  = (const float*)d_in[10];
    const float* proj_w2  = (const float*)d_in[11];
    const float* proj_b2  = (const float*)d_in[12];
    float*       out      = (float*)d_out;

    const int    M  = BB * SS;                       // 16384 activation rows
    const size_t HS = (size_t)BB * SS * DD;          // 8,388,608 floats

    float* ws   = (float*)d_ws;
    float* bh   = ws;                                // hidden state   [M,D]
    float* by   = ws + HS;                           // residual sum   [M,D]
    float* bq   = ws + 2 * HS;                       // Q / attn out   [M,D]
    float* bk   = ws + 3 * HS;                       // K              [M,D]
    float* bv   = ws + 4 * HS;                       // V              [M,D]
    float* bffn = ws + 2 * HS;                       // FFN hidden     [M,DFF] (reuses q/k/v + spare)
    float* bm   = ws + 6 * HS;                       // seq mean       [B,D]
    float* bp   = bm + (size_t)BB * DD;              // proj hidden    [B,D/2]

    // bf16 transposed-weight region (ushorts), 16B aligned
    unsigned short* wreg = (unsigned short*)(bp + (size_t)BB * (DD / 2));
    unsigned short* wt_embed = wreg;                              size_t o = (size_t)INN * DD;
    unsigned short* wt_qkvo[LL][4];
    for (int l = 0; l < LL; ++l)
        for (int j = 0; j < 4; ++j) { wt_qkvo[l][j] = wreg + o; o += (size_t)DD * DD; }
    unsigned short* wt_ffn1[LL];
    for (int l = 0; l < LL; ++l) { wt_ffn1[l] = wreg + o; o += (size_t)DD * DFF; }
    unsigned short* wt_ffn2[LL];
    for (int l = 0; l < LL; ++l) { wt_ffn2[l] = wreg + o; o += (size_t)DFF * DD; }
    unsigned short* wt_p1 = wreg + o; o += (size_t)DD * (DD / 2);
    unsigned short* wt_p2 = wreg + o;

    // ---- one-time weight convert+transpose (fp32 [K,N] -> bf16 [N,K])
    launch_wt(embed_w, wt_embed, INN, DD, stream);
    for (int l = 0; l < LL; ++l) {
        for (int j = 0; j < 4; ++j)
            launch_wt(qkvo_w + ((size_t)l * 4 + j) * DD * DD, wt_qkvo[l][j], DD, DD, stream);
        launch_wt(ffn_w1 + (size_t)l * DD * DFF, wt_ffn1[l], DD, DFF, stream);
        launch_wt(ffn_w2 + (size_t)l * DFF * DD, wt_ffn2[l], DFF, DD, stream);
    }
    launch_wt(proj_w1, wt_p1, DD, DD / 2, stream);
    launch_wt(proj_w2, wt_p2, DD / 2, NT, stream);

    // embed: h = x @ embed_w + embed_b
    launch_gemm_full(ACT_NONE, x, wt_embed, embed_b, nullptr, bh, M, DD, INN, stream);

    const int total = (int)HS;
    const int nDecompBlk = (total + 255) / 256;

    for (int l = 0; l < LL; ++l) {
        const float* bqv = qkvo_b + ((size_t)l * 4 + 0) * DD;
        const float* bkv = qkvo_b + ((size_t)l * 4 + 1) * DD;
        const float* bvv = qkvo_b + ((size_t)l * 4 + 2) * DD;
        const float* bov = qkvo_b + ((size_t)l * 4 + 3) * DD;

        launch_gemm_full(ACT_NONE, bh, wt_qkvo[l][0], bqv, nullptr, bq, M, DD, DD, stream);
        launch_gemm_full(ACT_NONE, bh, wt_qkvo[l][1], bkv, nullptr, bk, M, DD, DD, stream);
        launch_gemm_full(ACT_NONE, bh, wt_qkvo[l][2], bvv, nullptr, bv, M, DD, DD, stream);

        // circular autocorrelation + softmax + *V (in place over Q buffer)
        autocorr_softmax_kernel<<<BB * DD, 256, 0, stream>>>(bq, bk, bv, bq);

        // y = h + (attn @ Wo + bo)
        launch_gemm_full(ACT_NONE, bq, wt_qkvo[l][3], bov, bh, by, M, DD, DD, stream);
        decomp_kernel<<<nDecompBlk, 256, 0, stream>>>(by, bh, total);

        // FFN
        launch_gemm_full(ACT_GELU, bh, wt_ffn1[l], ffn_b1 + (size_t)l * DFF,
                         nullptr, bffn, M, DFF, DD, stream);
        launch_gemm_full(ACT_NONE, bffn, wt_ffn2[l], ffn_b2 + (size_t)l * DD,
                         bh, by, M, DD, DFF, stream);
        decomp_kernel<<<nDecompBlk, 256, 0, stream>>>(by, bh, total);
    }

    // head
    seqmean_kernel<<<(BB * DD + 255) / 256, 256, 0, stream>>>(bh, bm);
    launch_gemm_guard(ACT_RELU, bm, wt_p1, proj_b1, nullptr, bp, BB, DD / 2, DD, stream);
    launch_gemm_guard(ACT_NONE, bp, wt_p2, proj_b2, nullptr, out, BB, NT, DD / 2, stream);
}
